// Net_210311_7670811590823
// MI455X (gfx1250) — matrix operand order repaired and checked by reading, not compiled, not run on hardware
//
#include <hip/hip_runtime.h>

#define NNODES 65536
#define NEDGES 1048576
#define NB     64
#define NPG    1024
#define HD     128

typedef __attribute__((ext_vector_type(16))) __bf16 v16bf;
typedef __attribute__((ext_vector_type(2)))  __bf16 v2bf;
typedef __attribute__((ext_vector_type(8)))  float  v8f;

// ---------- helpers ----------
__device__ __forceinline__ unsigned pk_bf16(float a, float b) {
#if __has_builtin(__builtin_amdgcn_cvt_pk_bf16_f32)
    union { v2bf v; unsigned u; } r;
    r.v = __builtin_amdgcn_cvt_pk_bf16_f32(a, b);   // hw RNE packed convert
    return r.u;
#else
    // round-to-nearest (half away): +0x8000 then take high halves via v_perm_b32
    union { float f; unsigned u; } ua, ub;
    ua.f = a; ub.f = b;
    unsigned x = ua.u + 0x8000u;
    unsigned y = ub.u + 0x8000u;
    return __builtin_amdgcn_perm(y, x, 0x07060302u);  // {hi16(y), hi16(x)}
#endif
}

union FragU { v16bf v; unsigned u[8]; uint4 q[2]; };

// A fragment (16x32 bf16, ISA 7.12.2): this lane's 16 values are
// row = lane&15, K = kOff..kOff+7 and kOff+16..kOff+23 (kOff already
// includes +8 for lanes 16..31).
__device__ __forceinline__ v16bf load_a_frag(const float* rowPtr, int kOff) {
    FragU r;
    float4 a = *(const float4*)(rowPtr + kOff);
    float4 b = *(const float4*)(rowPtr + kOff + 4);
    float4 c = *(const float4*)(rowPtr + kOff + 16);
    float4 d = *(const float4*)(rowPtr + kOff + 20);
    r.u[0] = pk_bf16(a.x, a.y); r.u[1] = pk_bf16(a.z, a.w);
    r.u[2] = pk_bf16(b.x, b.y); r.u[3] = pk_bf16(b.z, b.w);
    r.u[4] = pk_bf16(c.x, c.y); r.u[5] = pk_bf16(c.z, c.w);
    r.u[6] = pk_bf16(d.x, d.y); r.u[7] = pk_bf16(d.z, d.w);
    return r.v;
}

// Load a pre-packed B fragment: 32 contiguous bytes per lane.
__device__ __forceinline__ v16bf load_b_packed(const unsigned* Wp, int nt, int kc,
                                               int lane) {
    const uint4* p = (const uint4*)(Wp + (((nt << 2) + kc) * 32 + lane) * 8);
    FragU r;
    r.q[0] = p[0];
    r.q[1] = p[1];
    return r.v;
}

// ---------- weight pre-pack: f32 [128,128] -> bf16 WMMA B fragments ----------
// Fragment layout: [nt(8)][kc(4)][lane(32)][8 dwords].
// B (32x16 bf16): column = lane&15; lanes 0-15 hold K=k0..k0+7,k0+16..k0+23,
// lanes 16-31 hold K=k0+8..k0+15,k0+24..k0+31 (matches A fragment order).
__global__ void pack_wfrag(const float* __restrict__ W, unsigned* __restrict__ out) {
    int tid = blockIdx.x * blockDim.x + threadIdx.x;   // 1024 threads
    if (tid >= 8 * 4 * 32) return;
    int lane = tid & 31;
    int kc = (tid >> 5) & 3;
    int nt = tid >> 7;
    int col = nt * 16 + (lane & 15);
    int ks = kc * 32 + (lane >> 4) * 8;
    unsigned* o = out + tid * 8;
#pragma unroll
    for (int j = 0; j < 8; ++j)
        o[j] = pk_bf16(W[(ks + 2 * j + (j >= 4 ? 8 : 0)) * HD + col], W[(ks + 2 * j + 1 + (j >= 4 ? 8 : 0)) * HD + col]);
}

// ---------- generic fill ----------
__global__ void fill_f32(float* p, float v, int n) {
    int i = blockIdx.x * blockDim.x + threadIdx.x;
    if (i < n) p[i] = v;
}

// ---------- layer 1: scatter (feature dim 4) ----------
__global__ void scatter4(const float* __restrict__ x, const int* __restrict__ src,
                         const int* __restrict__ dst, const float* __restrict__ emask,
                         float* agg4, int E) {
    int e = blockIdx.x * blockDim.x + threadIdx.x;
    if (e >= E) return;
    float m = emask[e];
    if (m == 0.0f) return;
    int s = src[e], d = dst[e];
    float4 v = ((const float4*)x)[s];
    atomicAdd(&agg4[d * 4 + 0], v.x * m);
    atomicAdd(&agg4[d * 4 + 1], v.y * m);
    atomicAdd(&agg4[d * 4 + 2], v.z * m);
    atomicAdd(&agg4[d * 4 + 3], v.w * m);
}

// ---------- layer 1: conv (K=4, memory bound -> scalar) ----------
__global__ void conv_k4(const float* __restrict__ agg4, const float* __restrict__ Wr,
                        const float* __restrict__ x, const float* __restrict__ Wl,
                        const float* __restrict__ bias, const float* __restrict__ nmask,
                        float* h) {
    int tid = blockIdx.x * blockDim.x + threadIdx.x;
    if (tid >= NNODES * HD) return;
    int i = tid >> 7, f = tid & 127;
    float acc = bias[f];
#pragma unroll
    for (int k = 0; k < 4; ++k)
        acc += agg4[i * 4 + k] * Wr[k * HD + f] + x[i * 4 + k] * Wl[k * HD + f];
    h[tid] = fmaxf(acc, 0.0f) * nmask[i];
}

// ---------- scatter for H=128 (wave per edge, float4 per lane) ----------
__global__ void scatter_h(const float* __restrict__ h, const int* __restrict__ src,
                          const int* __restrict__ dst, const float* __restrict__ emask,
                          float* agg, int E) {
    int tid = blockIdx.x * blockDim.x + threadIdx.x;
    int e = tid >> 5, lane = tid & 31;
    if (e >= E) return;
    float m = emask[e];
    if (m == 0.0f) return;
    int s = src[e], d = dst[e];
    float4 v = *(const float4*)(h + s * HD + lane * 4);
    float* p = agg + d * HD + lane * 4;
    atomicAdd(p + 0, v.x * m);
    atomicAdd(p + 1, v.y * m);
    atomicAdd(p + 2, v.z * m);
    atomicAdd(p + 3, v.w * m);
}

// ---------- WMMA conv: out = relu(agg@WrP + hin@WlP + b) * nmask ----------
// One wave per 16-row block; computes the full 16x128 output -> in-place safe.
// Weights arrive pre-packed as bf16 B fragments; B loads are software
// pipelined one WMMA ahead.
__global__ void gemm_conv(const float* __restrict__ agg, const unsigned* __restrict__ WrP,
                          const float* hin, const unsigned* __restrict__ WlP,
                          const float* __restrict__ bias, const float* __restrict__ nmask,
                          float* out) {
    int wave = threadIdx.x >> 5;
    int lane = threadIdx.x & 31;
    int rt = blockIdx.x * 8 + wave;             // row tile (16 rows)
    int kg = lane >> 4;                         // lane half
    int mrow = rt * 16 + (lane & 15);
    int colBase = lane & 15;

    v8f zero = {0.f, 0.f, 0.f, 0.f, 0.f, 0.f, 0.f, 0.f};
    v8f acc[8];
#pragma unroll
    for (int nt = 0; nt < 8; ++nt) acc[nt] = zero;

    const float* aggRow = agg + mrow * HD;
    const float* hRow = hin + mrow * HD;

#pragma unroll
    for (int kc = 0; kc < 4; ++kc) {
        int kOff = kc * 32 + kg * 8;
        v16bf aA = load_a_frag(aggRow, kOff);
        v16bf aH = load_a_frag(hRow, kOff);
        // 16 WMMAs: j<8 -> agg@Wr, j>=8 -> hin@Wl; B fragment j+1 is loaded
        // before WMMA j issues (double buffer), all indices compile-time.
        v16bf nxt = load_b_packed(WrP, 0, kc, lane);
#pragma unroll
        for (int j = 0; j < 16; ++j) {
            v16bf cur = nxt;
            if (j < 15) {
                int jn = j + 1;
                nxt = load_b_packed((jn < 8) ? WrP : WlP, jn & 7, kc, lane);
            }
            acc[j & 7] = __builtin_amdgcn_wmma_f32_16x16x32_bf16(
                false, (j < 8) ? aA : aH, false, cur, (short)0, acc[j & 7], false,
                false);
        }
    }

    int lm = kg * 8;  // C layout: VGPR v -> M = v + 8*(lane>=16), N = lane&15
#pragma unroll
    for (int nt = 0; nt < 8; ++nt) {
        int coln = nt * 16 + colBase;
        float bv = bias[coln];
#pragma unroll
        for (int v = 0; v < 8; ++v) {
            int row = rt * 16 + v + lm;
            float val = acc[nt][v] + bv;
            out[row * HD + coln] = fmaxf(val, 0.0f) * nmask[row];
        }
    }
}

// ---------- TopK pooling (one 1024-thread WG per graph) ----------
__global__ void topk_pool(float* h, const float* __restrict__ pw, float* nmask,
                          float ratio) {
    __shared__ float s[NPG];
    __shared__ int cnt;
    int g = blockIdx.x, t = threadIdx.x;
    int node = g * NPG + t;
    if (t == 0) cnt = 0;
    __syncthreads();

    float nm = nmask[node];
    float dot = 0.0f, nrm = 0.0f;
    const float* hr = h + node * HD;
    for (int k = 0; k < HD; ++k) {
        float wk = pw[k];
        dot += hr[k] * wk;
        nrm += wk * wk;
    }
    float sc = -__builtin_inff();
    if (nm > 0.0f) {
        sc = dot * rsqrtf(nrm);
        atomicAdd(&cnt, 1);
    }
    s[t] = sc;
    __syncthreads();

    int k = (int)ceilf(ratio * (float)cnt);
    int r = 0;
    for (int j = 0; j < NPG; ++j) {
        float sj = s[j];
        r += (sj > sc || (sj == sc && j < t)) ? 1 : 0;
    }
    float keep = (r < k && nm > 0.0f) ? 1.0f : 0.0f;
    float factor = (keep > 0.0f) ? tanhf(sc) : 0.0f;
    nmask[node] = keep;
    float* hw = h + node * HD;
    for (int kk = 0; kk < HD; ++kk) hw[kk] *= factor;
}

// ---------- edge mask update ----------
__global__ void emask_upd(float* emask, const int* __restrict__ src,
                          const int* __restrict__ dst, const float* __restrict__ nmask,
                          int E) {
    int e = blockIdx.x * blockDim.x + threadIdx.x;
    if (e >= E) return;
    emask[e] *= nmask[src[e]] * nmask[dst[e]];
}

// ---------- readout (max|mean per graph), accumulates into zsum ----------
__global__ void readout_add(const float* __restrict__ h, const float* __restrict__ nmask,
                            float* zsum) {
    int g = blockIdx.x, f = threadIdx.x;  // 128 threads
    float mx = -__builtin_inff(), sum = 0.0f, cnt = 0.0f;
    for (int n = 0; n < NPG; ++n) {
        float nm = nmask[g * NPG + n];
        float v = h[(g * NPG + n) * HD + f];
        if (nm > 0.0f) {
            mx = fmaxf(mx, v);
            sum += v;
            cnt += 1.0f;
        }
    }
    zsum[g * 256 + f] += mx;
    zsum[g * 256 + HD + f] += sum / cnt;
}

// ---------- MLP head ----------
__global__ void lin_relu(const float* __restrict__ in, const float* __restrict__ W,
                         const float* __restrict__ b, float* out, int rows, int fin,
                         int fout) {
    int tid = blockIdx.x * blockDim.x + threadIdx.x;
    if (tid >= rows * fout) return;
    int g = tid / fout, f = tid % fout;
    float acc = b[f];
    for (int k = 0; k < fin; ++k) acc += in[g * fin + k] * W[k * fout + f];
    out[tid] = fmaxf(acc, 0.0f);
}

__global__ void lin_logsoftmax(const float* __restrict__ in, const float* __restrict__ W,
                               const float* __restrict__ b, float* out, int rows) {
    int g = blockIdx.x * blockDim.x + threadIdx.x;
    if (g >= rows) return;
    float l[7];
#pragma unroll
    for (int c = 0; c < 7; ++c) {
        float acc = b[c];
        for (int k = 0; k < 64; ++k) acc += in[g * 64 + k] * W[k * 7 + c];
        l[c] = acc;
    }
    float m = l[0];
#pragma unroll
    for (int c = 1; c < 7; ++c) m = fmaxf(m, l[c]);
    float sum = 0.0f;
#pragma unroll
    for (int c = 0; c < 7; ++c) sum += expf(l[c] - m);
    float lse = logf(sum);
#pragma unroll
    for (int c = 0; c < 7; ++c) out[g * 7 + c] = l[c] - m - lse;
}

// ---------- host ----------
extern "C" void kernel_launch(void* const* d_in, const int* in_sizes, int n_in,
                              void* d_out, int out_size, void* d_ws, size_t ws_size,
                              hipStream_t stream) {
    const float* x    = (const float*)d_in[0];
    const int*   ei   = (const int*)d_in[1];
    const int*   src  = ei;
    const int*   dst  = ei + NEDGES;
    const float* w1r  = (const float*)d_in[2];
    const float* b1   = (const float*)d_in[3];
    const float* w1l  = (const float*)d_in[4];
    const float* w2r  = (const float*)d_in[5];
    const float* b2   = (const float*)d_in[6];
    const float* w2l  = (const float*)d_in[7];
    const float* w3r  = (const float*)d_in[8];
    const float* b3   = (const float*)d_in[9];
    const float* w3l  = (const float*)d_in[10];
    const float* p1w  = (const float*)d_in[11];
    const float* p2w  = (const float*)d_in[12];
    const float* wl1  = (const float*)d_in[13];
    const float* bl1  = (const float*)d_in[14];
    const float* wl2  = (const float*)d_in[15];
    const float* bl2  = (const float*)d_in[16];
    const float* wl3  = (const float*)d_in[17];
    const float* bl3  = (const float*)d_in[18];

    float* ws    = (float*)d_ws;
    float* h     = ws;                           // N*128
    float* agg   = h + (size_t)NNODES * HD;      // N*128 (layer1 uses first N*4)
    float* nmask = agg + (size_t)NNODES * HD;    // N
    float* emask = nmask + NNODES;               // E
    float* zsum  = emask + NEDGES;               // B*256
    float* t1    = zsum + NB * 256;              // B*128
    float* t2    = t1 + NB * HD;                 // B*64
    unsigned* wp = (unsigned*)(t2 + NB * 64);    // 4 x 8192 u32 packed weights
    unsigned* wp2r = wp;
    unsigned* wp2l = wp + 8192;
    unsigned* wp3r = wp + 16384;
    unsigned* wp3l = wp + 24576;

    const int TB = 256;
    auto blk = [](int n, int t) { return (n + t - 1) / t; };

    // init + one-time weight packing
    fill_f32<<<blk(NNODES, TB), TB, 0, stream>>>(nmask, 1.0f, NNODES);
    fill_f32<<<blk(NEDGES, TB), TB, 0, stream>>>(emask, 1.0f, NEDGES);
    fill_f32<<<blk(NB * 256, TB), TB, 0, stream>>>(zsum, 0.0f, NB * 256);
    fill_f32<<<blk(NNODES * 4, TB), TB, 0, stream>>>(agg, 0.0f, NNODES * 4);
    pack_wfrag<<<4, TB, 0, stream>>>(w2r, wp2r);
    pack_wfrag<<<4, TB, 0, stream>>>(w2l, wp2l);
    pack_wfrag<<<4, TB, 0, stream>>>(w3r, wp3r);
    pack_wfrag<<<4, TB, 0, stream>>>(w3l, wp3l);

    // ---- layer 1 ----
    scatter4<<<blk(NEDGES, TB), TB, 0, stream>>>(x, src, dst, emask, agg, NEDGES);
    conv_k4<<<blk(NNODES * HD, TB), TB, 0, stream>>>(agg, w1r, x, w1l, b1, nmask, h);
    topk_pool<<<NB, NPG, 0, stream>>>(h, p1w, nmask, 0.8f);
    emask_upd<<<blk(NEDGES, TB), TB, 0, stream>>>(emask, src, dst, nmask, NEDGES);
    readout_add<<<NB, HD, 0, stream>>>(h, nmask, zsum);

    // ---- layer 2 ----
    fill_f32<<<blk(NNODES * HD, TB), TB, 0, stream>>>(agg, 0.0f, NNODES * HD);
    scatter_h<<<blk(NEDGES * 32, TB), TB, 0, stream>>>(h, src, dst, emask, agg, NEDGES);
    gemm_conv<<<NNODES / (16 * 8), TB, 0, stream>>>(agg, wp2r, h, wp2l, b2, nmask, h);
    topk_pool<<<NB, NPG, 0, stream>>>(h, p2w, nmask, 0.8f);
    emask_upd<<<blk(NEDGES, TB), TB, 0, stream>>>(emask, src, dst, nmask, NEDGES);
    readout_add<<<NB, HD, 0, stream>>>(h, nmask, zsum);

    // ---- layer 3 (reuses p2w, matching the reference) ----
    fill_f32<<<blk(NNODES * HD, TB), TB, 0, stream>>>(agg, 0.0f, NNODES * HD);
    scatter_h<<<blk(NEDGES * 32, TB), TB, 0, stream>>>(h, src, dst, emask, agg, NEDGES);
    gemm_conv<<<NNODES / (16 * 8), TB, 0, stream>>>(agg, wp3r, h, wp3l, b3, nmask, h);
    topk_pool<<<NB, NPG, 0, stream>>>(h, p2w, nmask, 0.8f);
    emask_upd<<<blk(NEDGES, TB), TB, 0, stream>>>(emask, src, dst, nmask, NEDGES);
    readout_add<<<NB, HD, 0, stream>>>(h, nmask, zsum);

    // ---- MLP head ----
    lin_relu<<<blk(NB * HD, TB), TB, 0, stream>>>(zsum, wl1, bl1, t1, NB, 256, HD);
    lin_relu<<<blk(NB * 64, TB), TB, 0, stream>>>(t1, wl2, bl2, t2, NB, HD, 64);
    lin_logsoftmax<<<1, NB, 0, stream>>>(t2, wl3, bl3, (float*)d_out, NB);
}
